// MultiModalTransformer_39960375722579
// MI455X (gfx1250) — matrix operand order repaired and checked by reading, not compiled, not run on hardware
//
#include <hip/hip_runtime.h>

// ---------------------------------------------------------------------------
// MultiModalTransformer fused kernel for gfx1250 (MI455X, wave32, WMMA).
// Seq-len == 1  =>  softmax over a singleton axis == 1  =>  attn == V-proj.
// So the whole network is:  y = LN2(W2·relu(W1·LN1(Wo·(Wv·x+bv)+bo)+b1)+b2)
// per (batch-row, mode): four GEMMs driven by v_wmma_f32_16x16x32_bf16.
// ---------------------------------------------------------------------------

typedef __bf16 bf16;
typedef __attribute__((ext_vector_type(16))) __bf16 v16bf;
typedef __attribute__((ext_vector_type(8)))  __bf16 v8bf;
typedef __attribute__((ext_vector_type(8)))  float  v8f;

#define B_TOT  8192
#define M_MOD  20
#define E_DIM  256
#define FF_DIM 1024
#define ROWS   16
#define LDX    264    // bf16 stride for 256-wide activation tile
#define LDH    1032   // bf16 stride for 1024-wide hidden tile
#define LDC    260    // f32 stride for 256-wide f32 scratch

__device__ __forceinline__ bf16 f2bf(float f) { return (bf16)f; }

// One 16x16 output tile, accumulating A(16xK, LDS bf16) x W^T (W row-major
// [N_total x K] bf16 in global).  WMMA fragment layouts per CDNA5 ISA 7.12.2.
template <int K>
__device__ __forceinline__ v8f gemm_tile(const bf16* __restrict__ sA, int lda,
                                         const bf16* __restrict__ W, int n0,
                                         int lane)
{
    const int half = lane >> 4;
    const int mr   = lane & 15;         // A row for this lane
    const int nc   = lane & 15;         // B column (= weight row) for this lane
    const bf16* pa = sA + mr * lda + half * 8;
    const bf16* pb = W + (size_t)(n0 + nc) * K + half * 8;
    v8f acc = {};
#pragma unroll
    for (int kb = 0; kb < K; kb += 32) {
        v16bf a, b;
        // A 16x32 bf16: elems 0..7 -> K = kb + half*8 + 0..7
        //               elems 8..15 -> K = kb + 16 + half*8 + 0..7
        ((v8bf*)&a)[0] = *(const v8bf*)(pa + kb);
        ((v8bf*)&a)[1] = *(const v8bf*)(pa + kb + 16);
        // B 32x16 bf16: lane holds column nc, same K order as A
        ((v8bf*)&b)[0] = *(const v8bf*)(pb + kb);
        ((v8bf*)&b)[1] = *(const v8bf*)(pb + kb + 16);
        acc = __builtin_amdgcn_wmma_f32_16x16x32_bf16(
                  false, a, false, b, (short)0, acc, false, false);
    }
    return acc;
}

extern "C" __global__ __launch_bounds__(128)
void mmtx_fused(const float* __restrict__ value,
                const bf16*  __restrict__ wv, const float* __restrict__ bv,
                const bf16*  __restrict__ wo, const float* __restrict__ bo,
                const float* __restrict__ ln1g, const float* __restrict__ ln1b,
                const bf16*  __restrict__ w1, const float* __restrict__ b1,
                const bf16*  __restrict__ w2, const float* __restrict__ b2,
                const float* __restrict__ ln2g, const float* __restrict__ ln2b,
                float* __restrict__ out)
{
    __shared__ __align__(16) bf16  sX[ROWS * LDX];   // 256-wide activations
    __shared__ __align__(16) bf16  sH[ROWS * LDH];   // 1024-wide hidden
    __shared__ __align__(16) float sC[ROWS * LDC];   // f32 GEMM epilogue scratch
    __shared__ float sMu[ROWS], sRs[ROWS];

    const int m    = blockIdx.y;
    const int row0 = blockIdx.x * ROWS;
    const int tid  = threadIdx.x;
    const int lane = tid & 31;
    const int wid  = tid >> 5;
    const int half = lane >> 4;
    const int nc   = lane & 15;

    // ---- stage 16 rows of `value` into LDS as bf16 --------------------------
    {
        const float4* src = (const float4*)(value + (size_t)row0 * E_DIM);
#pragma unroll
        for (int it = 0; it < 8; ++it) {
            int idx = tid + it * 128;          // 1024 float4 chunks
            int r = idx >> 6, c4 = idx & 63;
            float4 v = src[(size_t)r * 64 + c4];
            bf16* d = sX + r * LDX + (c4 << 2);
            d[0] = f2bf(v.x); d[1] = f2bf(v.y); d[2] = f2bf(v.z); d[3] = f2bf(v.w);
        }
    }
    __syncthreads();

    // ---- GEMM1: V = X * Wv[m]^T + bv ---------------------------------------
    {
        const bf16* W = wv + (size_t)m * E_DIM * E_DIM;
        for (int t = wid; t < E_DIM / 16; t += 4) {
            int n0 = t * 16;
            v8f acc = gemm_tile<E_DIM>(sX, LDX, W, n0, lane);
            float bias = bv[m * E_DIM + n0 + nc];
#pragma unroll
            for (int v = 0; v < 8; ++v)
                sC[(v + half * 8) * LDC + n0 + nc] = acc[v] + bias;
        }
    }
    __syncthreads();
#pragma unroll
    for (int it = 0; it < 32; ++it) {          // f32 -> bf16 for next GEMM
        int idx = tid + it * 128;
        int r = idx >> 8, c = idx & 255;
        sX[r * LDX + c] = f2bf(sC[r * LDC + c]);
    }
    __syncthreads();

    // ---- GEMM2: X1 = V * Wo[m]^T + bo --------------------------------------
    {
        const bf16* W = wo + (size_t)m * E_DIM * E_DIM;
        for (int t = wid; t < E_DIM / 16; t += 4) {
            int n0 = t * 16;
            v8f acc = gemm_tile<E_DIM>(sX, LDX, W, n0, lane);
            float bias = bo[m * E_DIM + n0 + nc];
#pragma unroll
            for (int v = 0; v < 8; ++v)
                sC[(v + half * 8) * LDC + n0 + nc] = acc[v] + bias;
        }
    }
    __syncthreads();

    // ---- LayerNorm 1 -> bf16 in sX -----------------------------------------
    if (tid < ROWS) {
        const float4* p = (const float4*)(sC + tid * LDC);
        float s = 0.f, ss = 0.f;
#pragma unroll
        for (int i = 0; i < 64; ++i) {
            float4 v = p[i];
            s  += v.x + v.y + v.z + v.w;
            ss += v.x * v.x + v.y * v.y + v.z * v.z + v.w * v.w;
        }
        float mu  = s * (1.f / 256.f);
        float var = ss * (1.f / 256.f) - mu * mu;
        sMu[tid] = mu;
        sRs[tid] = rsqrtf(var + 1e-5f);
    }
    __syncthreads();
#pragma unroll
    for (int it = 0; it < 32; ++it) {
        int idx = tid + it * 128;
        int r = idx >> 8, c = idx & 255;
        float xv = (sC[r * LDC + c] - sMu[r]) * sRs[r] * ln1g[c] + ln1b[c];
        sX[r * LDX + c] = f2bf(xv);
    }
    __syncthreads();

    // ---- GEMM3: H = relu(LN1 * W1^T + b1), bf16 straight into sH -----------
    for (int t = wid; t < FF_DIM / 16; t += 4) {
        int n0 = t * 16;
        v8f acc = gemm_tile<E_DIM>(sX, LDX, w1, n0, lane);
        float bias = b1[n0 + nc];
#pragma unroll
        for (int v = 0; v < 8; ++v) {
            float hv = acc[v] + bias;
            hv = hv > 0.f ? hv : 0.f;
            sH[(v + half * 8) * LDH + n0 + nc] = f2bf(hv);
        }
    }
    __syncthreads();

    // ---- GEMM4: Y = H * W2^T + b2 (K = 1024) -------------------------------
    for (int t = wid; t < E_DIM / 16; t += 4) {
        int n0 = t * 16;
        v8f acc = gemm_tile<FF_DIM>(sH, LDH, w2, n0, lane);
        float bias = b2[n0 + nc];
#pragma unroll
        for (int v = 0; v < 8; ++v)
            sC[(v + half * 8) * LDC + n0 + nc] = acc[v] + bias;
    }
    __syncthreads();

    // ---- LayerNorm 2 + coalesced f32 store ---------------------------------
    if (tid < ROWS) {
        const float4* p = (const float4*)(sC + tid * LDC);
        float s = 0.f, ss = 0.f;
#pragma unroll
        for (int i = 0; i < 64; ++i) {
            float4 v = p[i];
            s  += v.x + v.y + v.z + v.w;
            ss += v.x * v.x + v.y * v.y + v.z * v.z + v.w * v.w;
        }
        float mu  = s * (1.f / 256.f);
        float var = ss * (1.f / 256.f) - mu * mu;
        sMu[tid] = mu;
        sRs[tid] = rsqrtf(var + 1e-5f);
    }
    __syncthreads();
#pragma unroll
    for (int it = 0; it < 32; ++it) {
        int idx = tid + it * 128;
        int r = idx >> 8, c = idx & 255;
        float yv = (sC[r * LDC + c] - sMu[r]) * sRs[r] * ln2g[c] + ln2b[c];
        out[((size_t)(row0 + r) * M_MOD + m) * E_DIM + c] = yv;
    }
}

// ---- f32 -> bf16 weight conversion (vec4, deterministic, runs every call) --
extern "C" __global__ void cvt_bf16_v4(const float* __restrict__ src,
                                       bf16* __restrict__ dst, int n4)
{
    int i = blockIdx.x * blockDim.x + threadIdx.x;
    if (i < n4) {
        float4 v = ((const float4*)src)[i];
        bf16* d = dst + (size_t)i * 4;
        d[0] = f2bf(v.x); d[1] = f2bf(v.y); d[2] = f2bf(v.z); d[3] = f2bf(v.w);
    }
}

extern "C" void kernel_launch(void* const* d_in, const int* in_sizes, int n_in,
                              void* d_out, int out_size, void* d_ws, size_t ws_size,
                              hipStream_t stream)
{
    (void)in_sizes; (void)n_in; (void)out_size; (void)ws_size;
    // setup_inputs order:
    // 0 query 1 key 2 value 3 Wq 4 bq 5 Wk 6 bk 7 Wv 8 bv 9 Wo 10 bo
    // 11 ln1_g 12 ln1_b 13 W1 14 b1 15 W2 16 b2 17 ln2_g 18 ln2_b
    // Wq/bq/Wk/bk are mathematically dead (softmax over singleton == 1).
    const float* value = (const float*)d_in[2];
    const float* Wv  = (const float*)d_in[7];  const float* bv = (const float*)d_in[8];
    const float* Wo  = (const float*)d_in[9];  const float* bo = (const float*)d_in[10];
    const float* l1g = (const float*)d_in[11]; const float* l1b = (const float*)d_in[12];
    const float* W1  = (const float*)d_in[13]; const float* b1 = (const float*)d_in[14];
    const float* W2  = (const float*)d_in[15]; const float* b2 = (const float*)d_in[16];
    const float* l2g = (const float*)d_in[17]; const float* l2b = (const float*)d_in[18];

    const size_t nWv = (size_t)M_MOD * E_DIM * E_DIM;   // 1,310,720
    const size_t nW1 = (size_t)FF_DIM * E_DIM;          //   262,144
    bf16* wv = (bf16*)d_ws;
    bf16* wo = wv + nWv;
    bf16* w1 = wo + nWv;
    bf16* w2 = w1 + nW1;   // total 3,145,728 bf16 = 6.3 MB of workspace

    const int thr = 256;
    cvt_bf16_v4<<<(int)((nWv / 4 + thr - 1) / thr), thr, 0, stream>>>(Wv, wv, (int)(nWv / 4));
    cvt_bf16_v4<<<(int)((nWv / 4 + thr - 1) / thr), thr, 0, stream>>>(Wo, wo, (int)(nWv / 4));
    cvt_bf16_v4<<<(int)((nW1 / 4 + thr - 1) / thr), thr, 0, stream>>>(W1, w1, (int)(nW1 / 4));
    cvt_bf16_v4<<<(int)((nW1 / 4 + thr - 1) / thr), thr, 0, stream>>>(W2, w2, (int)(nW1 / 4));

    dim3 grid(B_TOT / ROWS, M_MOD);
    mmtx_fused<<<grid, 128, 0, stream>>>(value,
                                         wv, bv, wo, bo,
                                         l1g, l1b,
                                         w1, b1, w2, b2,
                                         l2g, l2b,
                                         (float*)d_out);
}
